// IBR_38525856645311
// MI455X (gfx1250) — matrix operand order repaired and checked by reading, not compiled, not run on hardware
//
#include <hip/hip_runtime.h>
#include <hip/hip_bf16.h>
#include <math.h>

typedef __attribute__((ext_vector_type(16))) _Float16 v16h;
typedef __attribute__((ext_vector_type(8)))  float    v8f;

#define HW 12544      // 112*112
#define NB 4
#define NC 64
#define KK 49
#define NPIX_TOT (NB * NC * HW)

// ---------------------------------------------------------------------------
// Kernel 1: z[b,o,hw] = sum_c W_reduce[o,c] * x[b,c,hw]
// Block: 256 threads (8 waves). Tile: M=64 (all out ch), N=128 pixels, K=64.
// Each wave owns a 16-pixel N strip, loops 4 M-tiles, 2 K=32 chunks -> 8 WMMA.
// ---------------------------------------------------------------------------
__global__ __launch_bounds__(256) void k_gemm_reduce(
    const float* __restrict__ x, const float* __restrict__ w, float* __restrict__ z)
{
  __shared__ _Float16 lw[64 * 64];    // [M][K]  8 KB
  __shared__ _Float16 lx[64 * 128];   // [K][N] 16 KB
  const int b   = blockIdx.y;
  const int hw0 = blockIdx.x * 128;
  const int tid = threadIdx.x;

  for (int i = tid; i < 64 * 64; i += 256) lw[i] = (_Float16)w[i];
  const float* xb = x + b * NC * HW + hw0;
  for (int i = tid; i < 64 * 128; i += 256) {
    int c = i >> 7, n = i & 127;
    lx[i] = (_Float16)xb[c * HW + n];
  }
  __syncthreads();

  const int wave = tid >> 5, lane = tid & 31;
  const int nbase = wave * 16;
  const int mrow = lane & 15, lg = lane >> 4;
  const int ncol = lane & 15;

  v16h bf[2];
#pragma unroll
  for (int kc = 0; kc < 2; ++kc) {
    const int kb = kc * 32 + lg * 16;
#pragma unroll
    for (int i = 0; i < 16; ++i) bf[kc][i] = lx[(kb + i) * 128 + nbase + ncol];
  }

  float* zb = z + b * NC * HW + hw0 + nbase + ncol;
#pragma unroll
  for (int mt = 0; mt < 4; ++mt) {
    v8f acc = {};
#pragma unroll
    for (int kc = 0; kc < 2; ++kc) {
      v16h af;
      const int M = mt * 16 + mrow;
#pragma unroll
      for (int i = 0; i < 16; ++i) {
        const int K = kc * 32 + lg * 16 + i;
        af[i] = lw[M * 64 + K];
      }
      acc = __builtin_amdgcn_wmma_f32_16x16x32_f16(
          false, af, false, bf[kc], (short)0, acc, false, false);
    }
#pragma unroll
    for (int r = 0; r < 8; ++r) {
      const int M = mt * 16 + r + 8 * lg;
      zb[M * HW] = acc[r];
    }
  }
}

// ---------------------------------------------------------------------------
// Kernel 2: BN1 stats per channel over (B,H,W); fold gamma/beta into sc/sh.
// One block per channel.
// ---------------------------------------------------------------------------
__global__ __launch_bounds__(256) void k_bn1_stats(
    const float* __restrict__ z, const float* __restrict__ gamma,
    const float* __restrict__ beta, float* __restrict__ sc, float* __restrict__ sh)
{
  __shared__ float s1[256], s2[256];
  const int c = blockIdx.x, tid = threadIdx.x;
  float sum = 0.f, sq = 0.f;
  for (int b = 0; b < NB; ++b) {
    const float* p = z + (b * NC + c) * HW;
    for (int i = tid; i < HW; i += 256) { float v = p[i]; sum += v; sq += v * v; }
  }
  s1[tid] = sum; s2[tid] = sq; __syncthreads();
  for (int s = 128; s > 0; s >>= 1) {
    if (tid < s) { s1[tid] += s1[tid + s]; s2[tid] += s2[tid + s]; }
    __syncthreads();
  }
  if (tid == 0) {
    const float n = (float)(NB * HW);
    const float mean = s1[0] / n;
    const float var  = s2[0] / n - mean * mean;
    const float rstd = rsqrtf(var + 1e-5f);
    const float g = gamma[c] * rstd;
    sc[c] = g;
    sh[c] = beta[c] - mean * g;
  }
}

// ---------------------------------------------------------------------------
// Kernel 3: kern[b,p,hw] = sum_c W_span[p,c] * relu(z*sc + sh)
// Same WMMA structure; M padded 49 -> 64 with zero rows; BN1+ReLU applied on
// the fly while converting the z tile to f16 in LDS.
// ---------------------------------------------------------------------------
__global__ __launch_bounds__(256) void k_gemm_span(
    const float* __restrict__ z, const float* __restrict__ wspan,
    const float* __restrict__ sc, const float* __restrict__ sh,
    float* __restrict__ kern)
{
  __shared__ _Float16 lw[64 * 64];
  __shared__ _Float16 ly[64 * 128];
  __shared__ float ssc[64], ssh[64];
  const int b   = blockIdx.y;
  const int hw0 = blockIdx.x * 128;
  const int tid = threadIdx.x;

  if (tid < 64) { ssc[tid] = sc[tid]; ssh[tid] = sh[tid]; }
  for (int i = tid; i < 64 * 64; i += 256)
    lw[i] = (i < KK * 64) ? (_Float16)wspan[i] : (_Float16)0.0f;
  __syncthreads();

  const float* zb = z + b * NC * HW + hw0;
  for (int i = tid; i < 64 * 128; i += 256) {
    int c = i >> 7, n = i & 127;
    float v = zb[c * HW + n] * ssc[c] + ssh[c];
    ly[i] = (_Float16)(v > 0.f ? v : 0.f);
  }
  __syncthreads();

  const int wave = tid >> 5, lane = tid & 31;
  const int nbase = wave * 16;
  const int mrow = lane & 15, lg = lane >> 4;
  const int ncol = lane & 15;

  v16h bf[2];
#pragma unroll
  for (int kc = 0; kc < 2; ++kc) {
    const int kb = kc * 32 + lg * 16;
#pragma unroll
    for (int i = 0; i < 16; ++i) bf[kc][i] = ly[(kb + i) * 128 + nbase + ncol];
  }

  float* kb = kern + b * KK * HW + hw0 + nbase + ncol;
#pragma unroll
  for (int mt = 0; mt < 4; ++mt) {
    v8f acc = {};
#pragma unroll
    for (int kc = 0; kc < 2; ++kc) {
      v16h af;
      const int M = mt * 16 + mrow;
#pragma unroll
      for (int i = 0; i < 16; ++i) {
        const int K = kc * 32 + lg * 16 + i;
        af[i] = lw[M * 64 + K];
      }
      acc = __builtin_amdgcn_wmma_f32_16x16x32_f16(
          false, af, false, bf[kc], (short)0, acc, false, false);
    }
#pragma unroll
    for (int r = 0; r < 8; ++r) {
      const int M = mt * 16 + r + 8 * lg;
      if (M < KK) kb[M * HW] = acc[r];
    }
  }
}

// ---------------------------------------------------------------------------
// Kernel 4: involution. Block = one (b, c, 16x16 pixel tile); x halo in LDS.
// Writes raw output to d_out and deterministic per-block BN2 partials.
// ---------------------------------------------------------------------------
__global__ __launch_bounds__(256) void k_invol(
    const float* __restrict__ x, const float* __restrict__ kern,
    float* __restrict__ outraw, float* __restrict__ part)
{
  __shared__ float xs[22 * 22];
  __shared__ float r1[256], r2[256];
  const int tx = blockIdx.x, ty = blockIdx.y;      // 7x7 pixel tiles
  const int bc = blockIdx.z;                        // b*64 + c
  const int b = bc >> 6, c = bc & 63;
  const int tid = threadIdx.x;
  const int lw_ = tid & 15, lh = tid >> 4;
  const int h0 = ty * 16, w0 = tx * 16;

  const float* xp = x + (b * NC + c) * HW;
  for (int i = tid; i < 22 * 22; i += 256) {
    const int hh = h0 - 3 + i / 22, ww = w0 - 3 + i % 22;
    xs[i] = (hh >= 0 && hh < 112 && ww >= 0 && ww < 112) ? xp[hh * 112 + ww] : 0.f;
  }
  __syncthreads();

  const int h = h0 + lh, w = w0 + lw_;
  const int hw = h * 112 + w;
  const float* kp = kern + b * KK * HW + hw;
  float acc = 0.f;
#pragma unroll
  for (int p = 0; p < KK; ++p) {
    const int kh = p / 7, kw = p % 7;
    acc += xs[(lh + kh) * 22 + (lw_ + kw)] * kp[p * HW];
  }
  outraw[(b * NC + c) * HW + hw] = acc;

  r1[tid] = acc; r2[tid] = acc * acc; __syncthreads();
  for (int s = 128; s > 0; s >>= 1) {
    if (tid < s) { r1[tid] += r1[tid + s]; r2[tid] += r2[tid + s]; }
    __syncthreads();
  }
  if (tid == 0) {
    const int tile = (b * 49 + ty * 7 + tx);       // 196 tiles per channel
    part[(c * 196 + tile) * 2 + 0] = r1[0];
    part[(c * 196 + tile) * 2 + 1] = r2[0];
  }
}

// ---------------------------------------------------------------------------
// Kernel 5: finalize BN2 stats -> per-channel scale/shift (fold gamma/beta).
// ---------------------------------------------------------------------------
__global__ __launch_bounds__(64) void k_bn2_finalize(
    const float* __restrict__ part, const float* __restrict__ gamma,
    const float* __restrict__ beta, float* __restrict__ sc2, float* __restrict__ sh2)
{
  const int c = threadIdx.x;
  float sum = 0.f, sq = 0.f;
  for (int t = 0; t < 196; ++t) {
    sum += part[(c * 196 + t) * 2 + 0];
    sq  += part[(c * 196 + t) * 2 + 1];
  }
  const float n = (float)(NB * HW);
  const float mean = sum / n;
  const float var  = sq / n - mean * mean;
  const float rstd = rsqrtf(var + 1e-3f);
  const float g = gamma[c] * rstd;
  sc2[c] = g;
  sh2[c] = beta[c] - mean * g;
}

// ---------------------------------------------------------------------------
// Kernel 6: apply BN2 + PReLU in place on d_out.
// ---------------------------------------------------------------------------
__global__ __launch_bounds__(256) void k_bn2_prelu(
    float* __restrict__ out, const float* __restrict__ sc2,
    const float* __restrict__ sh2, const float* __restrict__ alpha)
{
  const int i = blockIdx.x * 256 + threadIdx.x;
  if (i >= NPIX_TOT) return;
  const int c = (i / HW) & 63;
  const float v = out[i] * sc2[c] + sh2[c];
  out[i] = v > 0.f ? v : alpha[c] * v;
}

// ---------------------------------------------------------------------------
extern "C" void kernel_launch(void* const* d_in, const int* in_sizes, int n_in,
                              void* d_out, int out_size, void* d_ws, size_t ws_size,
                              hipStream_t stream) {
  const float* x        = (const float*)d_in[0];
  const float* w_reduce = (const float*)d_in[1];
  const float* bn1_g    = (const float*)d_in[2];
  const float* bn1_b    = (const float*)d_in[3];
  const float* w_span   = (const float*)d_in[4];
  const float* bn2_g    = (const float*)d_in[5];
  const float* bn2_b    = (const float*)d_in[6];
  const float* alpha    = (const float*)d_in[7];
  float* out = (float*)d_out;

  float* ws = (float*)d_ws;
  float* z    = ws;                              // 4*64*12544 = 3,211,264
  float* kern = z + (size_t)NB * NC * HW;        // 4*49*12544 = 2,458,624
  float* sc1  = kern + (size_t)NB * KK * HW;     // 64
  float* sh1  = sc1 + 64;                        // 64
  float* part = sh1 + 64;                        // 64*196*2 = 25,088
  float* sc2  = part + 64 * 196 * 2;             // 64
  float* sh2  = sc2 + 64;                        // 64

  dim3 gemm_grid(HW / 128, NB);                  // (98, 4)
  k_gemm_reduce<<<gemm_grid, 256, 0, stream>>>(x, w_reduce, z);
  k_bn1_stats<<<NC, 256, 0, stream>>>(z, bn1_g, bn1_b, sc1, sh1);
  k_gemm_span<<<gemm_grid, 256, 0, stream>>>(z, w_span, sc1, sh1, kern);
  dim3 inv_grid(7, 7, NB * NC);                  // 16x16 pixel tiles
  k_invol<<<inv_grid, 256, 0, stream>>>(x, kern, out, part);
  k_bn2_finalize<<<1, 64, 0, stream>>>(part, bn2_g, bn2_b, sc2, sh2);
  k_bn2_prelu<<<(NPIX_TOT + 255) / 256, 256, 0, stream>>>(out, sc2, sh2, alpha);
}
